// SpatialAttentionModule_25623774888590
// MI455X (gfx1250) — matrix operand order repaired and checked by reading, not compiled, not run on hardware
//
#include <hip/hip_runtime.h>
#include <hip/hip_bf16.h>
#include <math.h>

// Problem constants (from reference)
#define BB   32
#define TT   32
#define FEAT 49
#define LL   (TT*FEAT)   // 1568
#define DD   3072        // D_IN
#define HH   512
#define TWOH 1024
#define MBLK 64          // rows of L per block
#define NLT  25          // ceil(1568/64)
#define KPTOT (DD/2)     // 1536 bf16-pairs along K
#define NK   (DD/32)     // 96 K-steps

typedef __attribute__((ext_vector_type(16))) __bf16   v16bf;
typedef __attribute__((ext_vector_type(8)))  float    v8f;
typedef __attribute__((ext_vector_type(4)))  float    v4f;
typedef __attribute__((ext_vector_type(8)))  unsigned v8u;
typedef __attribute__((ext_vector_type(4)))  unsigned v4u;
typedef __attribute__((ext_vector_type(8)))  int      v8i;
typedef __attribute__((ext_vector_type(4)))  int      v4i;

// cheap fp32 -> bf16 pair pack (round-to-nearest)
__device__ __forceinline__ unsigned pack2(float lo, float hi) {
    unsigned ul = __builtin_bit_cast(unsigned, lo) + 0x8000u;
    unsigned uh = __builtin_bit_cast(unsigned, hi) + 0x8000u;
    return (ul >> 16) | (uh & 0xFFFF0000u);
}
// cheap tanh: (e^2x - 1)/(e^2x + 1), clamped so exp can't overflow
__device__ __forceinline__ float fast_tanh(float x) {
    x = fminf(fmaxf(x, -15.f), 15.f);
    float e = __expf(2.f * x);
    return (e - 1.f) / (e + 1.f);
}

// ---------------------------------------------------------------------------
// Kernel 0: pre-pack Wa (fp32 [DD,HH]) into TRANSPOSED bf16-pair dwords:
// WapkT[n][kp] = {Wa[2kp][n], Wa[2kp+1][n]},  [512][1536] dwords = 3MB.
// ---------------------------------------------------------------------------
__global__ __launch_bounds__(256)
void wapack_kernel(const float* __restrict__ Wa, unsigned* __restrict__ WapkT) {
    const int idx = blockIdx.x * 256 + threadIdx.x;  // over HH*KPTOT
    const int n  = idx / KPTOT;
    const int kp = idx - n * KPTOT;
    WapkT[idx] = pack2(Wa[(size_t)(2 * kp) * HH + n],
                       Wa[(size_t)(2 * kp + 1) * HH + n]);
}

// ---------------------------------------------------------------------------
// Kernel 1: u[b][h] = hidden_text[b] @ Ua + ba[h]
// ---------------------------------------------------------------------------
__global__ __launch_bounds__(512)
void uh_kernel(const float* __restrict__ ht, const float* __restrict__ Ua,
               const float* __restrict__ ba, float* __restrict__ u) {
    const int b = blockIdx.x;
    const int h = threadIdx.x;
    float acc = ba[h];
    const float* __restrict__ hrow = ht + (size_t)b * TWOH;
    #pragma unroll 8
    for (int k = 0; k < TWOH; ++k)
        acc = fmaf(hrow[k], Ua[(size_t)k * HH + h], acc);
    u[(size_t)b * HH + h] = acc;
}

// ---------------------------------------------------------------------------
// Kernel 2: fused scores, software-pipelined double buffer.
// Block = 256 threads (8 waves), 64 L-rows, full N=512.
// Per K-step: TDM loads the B tile into LDS (TENSORcnt), A tile is staged
// manually (fp32->bf16), both for step i+1 issued *under* step i's WMMAs.
// ---------------------------------------------------------------------------
__global__ __launch_bounds__(256)
void scores_kernel(const float* __restrict__ frames, const unsigned* __restrict__ WapkT,
                   const float* __restrict__ Va, const float* __restrict__ u,
                   float* __restrict__ scores) {
    __shared__ unsigned Asp[2][MBLK][20];  // 64 rows x 16 k-pairs (80B pitch)
    __shared__ unsigned Bsw[2][HH][20];    // 512 n x 16 k-pairs; TDM pads pitch
    __shared__ float    sred[MBLK];

    const int tid   = threadIdx.x;
    const int lane  = tid & 31;
    const int wv    = tid >> 5;            // 0..7 -> N base = wv*64
    const int b     = blockIdx.x / NLT;
    const int l0    = (blockIdx.x % NLT) * MBLK;
    const int rem   = (LL - l0 < MBLK) ? (LL - l0) : MBLK;  // 64 or 32 (tail)
    const int lhalf = (lane >> 4) & 1;
    const int l15   = lane & 15;

    v8f acc[4][4] = {};  // [m-tile][n-tile] fp32 accumulators

    const float* __restrict__ fbase = frames + ((size_t)b * LL + l0) * DD;
    const unsigned ldsB0 = (unsigned)(uintptr_t)&Bsw[0][0][0];
    const unsigned ldsB1 = (unsigned)(uintptr_t)&Bsw[1][0][0];

    // D# group1 (constant): data_size=4B(2<<16), pad_enable(1<<20),
    // pad_interval=16dw(3<<22), pad_amount=4dw(3<<25);
    // tensor_dim0=1536, tensor_dim1=512, tile_dim0=16, tile_dim1=512,
    // tensor_dim0_stride=1536 dwords.
    const v8i g1 = { (int)0x06D20000u,
                     (int)((KPTOT & 0xFFFF) << 16),
                     (int)((KPTOT >> 16) | ((HH & 0xFFFF) << 16)),
                     (int)((HH >> 16) | (16 << 16)),
                     (int)HH,
                     (int)KPTOT,
                     0, 0 };
    const v4i gz = { 0, 0, 0, 0 };
#if __has_include(<hip/amd_detail/amd_gfx1250_TDM.h>)
    const v8i gz8 = { 0, 0, 0, 0, 0, 0, 0, 0 };
#endif

    // stage(): issue TDM for B tile + stage A tile (bf16 pack) into buffer bufi
    auto stage = [&](int k0, int bufi) {
        if (wv == 0) {
            unsigned long long ga =
                (unsigned long long)(uintptr_t)WapkT + (unsigned long long)k0 * 2ull;
            v4u g0 = { 1u, bufi ? ldsB1 : ldsB0, (unsigned)ga,
                       (unsigned)(ga >> 32) | 0x80000000u };   // type=2 ("image")
#if __has_include(<hip/amd_detail/amd_gfx1250_TDM.h>)
            __builtin_amdgcn_tensor_load_to_lds(g0, g1, gz, gz, gz8, 0);
#else
            __builtin_amdgcn_tensor_load_to_lds(g0, g1, gz, gz, 0);
#endif
        }
        #pragma unroll
        for (int j = 0; j < 2; ++j) {
            int e   = tid + j * 256;             // 0..511 : 64 rows x 8 float4
            int row = e >> 3;
            int q   = e & 7;
            int gr  = (row < rem) ? row : (rem - 1);  // clamp tail, keep EXEC full
            v4f f = __builtin_nontemporal_load(
                        (const v4f*)(fbase + (size_t)gr * DD + k0 + q * 4));
            Asp[bufi][row][2 * q]     = pack2(f.x, f.y);
            Asp[bufi][row][2 * q + 1] = pack2(f.z, f.w);
        }
        if (k0 + 32 < DD) {   // reach one more step ahead
            int pr = tid >> 2;
            int pc = (tid & 3) * 8;
            __builtin_prefetch(fbase + (size_t)((pr < rem) ? pr : rem - 1) * DD
                                     + (k0 + 32) + pc, 0, 1);
        }
    };

    stage(0, 0);   // prologue

    for (int i = 0; i < NK; ++i) {
        const int buf = i & 1;
        if (wv == 0) __builtin_amdgcn_s_wait_tensorcnt(0);  // TDM(i) landed
        __syncthreads();   // publish TDM(i) + A-stores(i); retire compute(i-1)
        if (i + 1 < NK) stage((i + 1) * 32, buf ^ 1);       // next tile in flight

        // ---- compute from buffer `buf` ----
        const unsigned (*As)[20] = Asp[buf];
        const unsigned (*Bs)[20] = Bsw[buf];

        v16bf av[4];
        #pragma unroll
        for (int mt = 0; mt < 4; ++mt) {
            const unsigned* ar = &As[mt * 16 + l15][0];
            uint4 lo = *(const uint4*)(ar + lhalf * 4);
            uint4 hi = *(const uint4*)(ar + 8 + lhalf * 4);
            v8u au = {lo.x, lo.y, lo.z, lo.w, hi.x, hi.y, hi.z, hi.w};
            av[mt] = __builtin_bit_cast(v16bf, au);
        }
        #pragma unroll
        for (int nt = 0; nt < 4; ++nt) {
            const unsigned* br = &Bs[wv * 64 + nt * 16 + l15][0];
            uint4 blo = *(const uint4*)(br + lhalf * 4);
            uint4 bhi = *(const uint4*)(br + 8 + lhalf * 4);
            v8u bu = {blo.x, blo.y, blo.z, blo.w, bhi.x, bhi.y, bhi.z, bhi.w};
            v16bf bv = __builtin_bit_cast(v16bf, bu);
            #pragma unroll
            for (int mt = 0; mt < 4; ++mt) {
                acc[mt][nt] = __builtin_amdgcn_wmma_f32_16x16x32_bf16(
                    false, av[mt], false, bv, (short)0, acc[mt][nt], false, false);
            }
        }
    }

    // ---- epilogue: Va[n]*tanh(acc + u[b][n]), reduce over n ----
    float va_n[4], u_n[4];
    #pragma unroll
    for (int nt = 0; nt < 4; ++nt) {
        int n = wv * 64 + nt * 16 + l15;
        va_n[nt] = Va[n];
        u_n[nt]  = u[(size_t)b * HH + n];
    }
    float part[4][8];
    #pragma unroll
    for (int mt = 0; mt < 4; ++mt)
        #pragma unroll
        for (int r = 0; r < 8; ++r) {
            float s = 0.f;
            #pragma unroll
            for (int nt = 0; nt < 4; ++nt)
                s = fmaf(va_n[nt], fast_tanh(acc[mt][nt][r] + u_n[nt]), s);
            part[mt][r] = s;
        }
    #pragma unroll
    for (int off = 1; off < 16; off <<= 1)
        #pragma unroll
        for (int mt = 0; mt < 4; ++mt)
            #pragma unroll
            for (int r = 0; r < 8; ++r)
                part[mt][r] += __shfl_xor(part[mt][r], off, 32);

    __syncthreads();
    if (tid < MBLK) sred[tid] = 0.f;
    __syncthreads();
    if (l15 == 0) {
        #pragma unroll
        for (int mt = 0; mt < 4; ++mt)
            #pragma unroll
            for (int r = 0; r < 8; ++r)
                atomicAdd(&sred[mt * 16 + r + lhalf * 8], part[mt][r]);
    }
    __syncthreads();
    if (tid < rem) scores[(size_t)b * LL + l0 + tid] = sred[tid];
}

// ---------------------------------------------------------------------------
// Kernel 3: softmax over L=1568, in place, one block per b
// ---------------------------------------------------------------------------
__global__ __launch_bounds__(256)
void softmax_kernel(float* __restrict__ s) {
    __shared__ float red[256];
    const int b = blockIdx.x, tid = threadIdx.x;
    float* __restrict__ row = s + (size_t)b * LL;

    float m = -INFINITY;
    for (int l = tid; l < LL; l += 256) m = fmaxf(m, row[l]);
    red[tid] = m; __syncthreads();
    for (int o = 128; o > 0; o >>= 1) {
        if (tid < o) red[tid] = fmaxf(red[tid], red[tid + o]);
        __syncthreads();
    }
    const float mx = red[0]; __syncthreads();

    float sm = 0.f;
    for (int l = tid; l < LL; l += 256) {
        float e = __expf(row[l] - mx);
        row[l] = e; sm += e;
    }
    red[tid] = sm; __syncthreads();
    for (int o = 128; o > 0; o >>= 1) {
        if (tid < o) red[tid] += red[tid + o];
        __syncthreads();
    }
    const float inv = 1.f / red[0]; __syncthreads();
    for (int l = tid; l < LL; l += 256) row[l] *= inv;
}

// ---------------------------------------------------------------------------
// Kernel 4: ht_sum[b][d] = sum_l att[b][l]*frames[b][l][d]  (streaming, fp32)
// Non-temporal frames loads: 616MB read-once, keep L2 for the weights.
// ---------------------------------------------------------------------------
__global__ __launch_bounds__(256)
void pool_kernel(const float* __restrict__ frames, const float* __restrict__ att,
                 float* __restrict__ out) {
    const int b  = blockIdx.z;
    const int d  = blockIdx.x * 1024 + threadIdx.x * 4;
    const int l0 = blockIdx.y * (LL / 4);
    const float*  __restrict__ a = att + (size_t)b * LL + l0;

    v4f acc = { 0.f, 0.f, 0.f, 0.f };
    for (int i = 0; i < LL / 4; ++i) {
        const float w = a[i];
        v4f f = __builtin_nontemporal_load(
                    (const v4f*)(frames + ((size_t)b * LL + l0 + i) * DD + d));
        acc += w * f;
    }
    float* o = out + (size_t)b * DD + d;
    atomicAdd(o + 0, acc.x);
    atomicAdd(o + 1, acc.y);
    atomicAdd(o + 2, acc.z);
    atomicAdd(o + 3, acc.w);
}

// ---------------------------------------------------------------------------
extern "C" void kernel_launch(void* const* d_in, const int* in_sizes, int n_in,
                              void* d_out, int out_size, void* d_ws, size_t ws_size,
                              hipStream_t stream) {
    const float* frames = (const float*)d_in[0];  // [B,T,FEAT,D_IN]
    const float* ht     = (const float*)d_in[1];  // [B,2H]
    const float* Wa     = (const float*)d_in[2];  // [D_IN,H]
    const float* Ua     = (const float*)d_in[3];  // [2H,H]
    const float* Va     = (const float*)d_in[4];  // [H]
    const float* ba     = (const float*)d_in[5];  // [1,1,H]
    float* out = (float*)d_out;                   // [B,D_IN]

    float*    u      = (float*)d_ws;                          // [B,H]
    float*    scores = u + (size_t)BB * HH;                   // [B,L]
    unsigned* WapkT  = (unsigned*)(scores + (size_t)BB * LL); // [HH][KPTOT]

    wapack_kernel<<<HH * KPTOT / 256, 256, 0, stream>>>(Wa, WapkT);
    uh_kernel<<<BB, 512, 0, stream>>>(ht, Ua, ba, u);
    scores_kernel<<<BB * NLT, 256, 0, stream>>>(frames, WapkT, Va, u, scores);
    softmax_kernel<<<BB, 256, 0, stream>>>(scores);
    hipMemsetAsync(out, 0, (size_t)out_size * sizeof(float), stream);
    pool_kernel<<<dim3(DD / 1024, 4, BB), 256, 0, stream>>>(frames, scores, out);
}
